// CrossAttention_65541200937572
// MI455X (gfx1250) — matrix operand order repaired and checked by reading, not compiled, not run on hardware
//
#include <hip/hip_runtime.h>

// ---------------------------------------------------------------------------
// CrossAttention forward for MI455X (gfx1250).
//   - one-time f32->bf16 conversion of activations & weights
//   - bf16 WMMA GEMM for Q/K/V projections (V stored transposed)
//   - flash attention with TDM (tensor_load_to_lds) double-buffered K/V tiles
// B=2, LQ=LK=2048, HIDDEN=1024, 16 heads, head_dim=64.
// ---------------------------------------------------------------------------

typedef __attribute__((ext_vector_type(16))) __bf16 v16bf;
typedef __attribute__((ext_vector_type(8)))  __bf16 v8bf;
typedef __attribute__((ext_vector_type(4)))  __bf16 v4bf;
typedef __attribute__((ext_vector_type(8)))  float  v8f;
typedef __attribute__((ext_vector_type(4)))  float  v4f;
typedef __attribute__((ext_vector_type(4)))  unsigned int v4u;
typedef __attribute__((ext_vector_type(8)))  unsigned int v8u;

#define HIDDEN 1024
#define NH 16
#define HD 64
#define BATCH 2
#define SEQ 2048
#define HEAD_ELEMS (SEQ * HD)                 // 131072 elements per (b,h)
#define QKV_ELEMS  (BATCH * NH * HEAD_ELEMS)  // 4,194,304 elements per tensor

static __device__ __forceinline__ v8f wmma_bf16(v16bf a, v16bf b, v8f c) {
  // D = A(16x32 bf16) x B(32x16 bf16) + C(16x16 f32)
  return __builtin_amdgcn_wmma_f32_16x16x32_bf16(
      /*neg_a=*/false, a, /*neg_b=*/false, b,
      /*c_mod=*/(short)0, c, /*reuse_a=*/false, /*reuse_b=*/false);
}

// ---------------------------------------------------------------------------
// Tensor Data Mover: 2D tile load Global -> LDS.
// D# group0: count=1 | lds_addr | global_addr[56:0] | type=2.
// D# group1 (constant per tensor shape): data_size=1 (2B), tensor_dim0/1,
// tile_dim0/1, tensor_dim0_stride. Groups 2/3 zero (dims unused).
// Tracked by TENSORcnt; drained with s_wait_tensorcnt.
// ---------------------------------------------------------------------------
static __device__ __forceinline__ void tdm_load_2d(unsigned lds_addr,
                                                   const void* gptr, v8u g1) {
  const unsigned long long ga = (unsigned long long)(uintptr_t)gptr;
  v4u g0;
  g0[0] = 1u;                                  // count=1, user descriptor
  g0[1] = lds_addr;                            // LDS byte address
  g0[2] = (unsigned)ga;                        // global_addr[31:0]
  g0[3] = (unsigned)(ga >> 32) | (2u << 30);   // global_addr[56:32] | type=2
  const v4u gz = {0u, 0u, 0u, 0u};
  asm volatile("tensor_load_to_lds %0, %1, %2, %3"
               :: "s"(g0), "s"(g1), "s"(gz), "s"(gz)
               : "memory");
}

// group1 pack helper (bit positions per ISA 8.4):
//   d0=[ws_mask|data_size<<16]  d1=[abar|tdim0.lo<<16]  d2=[tdim0.hi|tdim1.lo<<16]
//   d3=[tdim1.hi|tile0<<16]     d4=[tile1|tile2<<16]    d5=stride0.lo
//   d6=[stride0.hi|stride1.lo<<16] d7=stride1.hi
#define TDM_G1(tdim0, tdim1, tile0, tile1, stride0)                         \
  ((v8u){ (1u << 16),                                                        \
          ((unsigned)(tdim0) & 0xFFFFu) << 16,                               \
          (((unsigned)(tdim0) >> 16) & 0xFFFFu) |                            \
              (((unsigned)(tdim1) & 0xFFFFu) << 16),                         \
          (((unsigned)(tdim1) >> 16) & 0xFFFFu) | ((unsigned)(tile0) << 16), \
          (unsigned)(tile1),                                                 \
          (unsigned)(stride0),                                               \
          0u, 0u })

// ---------------------------------------------------------------------------
// Elementwise f32 -> bf16 (vector x4). n4 = n/4.
// ---------------------------------------------------------------------------
__global__ __launch_bounds__(256)
void cvt_f32_bf16_kernel(const float* __restrict__ src, __bf16* __restrict__ dst,
                         int n4) {
  const int i = blockIdx.x * blockDim.x + threadIdx.x;
  if (i < n4) {
    const v4f f = ((const v4f*)src)[i];
    v4bf o;
#pragma unroll
    for (int e = 0; e < 4; ++e) o[e] = (__bf16)f[e];
    ((v4bf*)dst)[i] = o;
  }
}

// ---------------------------------------------------------------------------
// Projection: Y = (Xb @ Wb^T + bias) * scale  (all-bf16 operands, f32 acc).
// One wave computes a 16x32 output tile (one A fragment feeds two WMMAs).
//   ls=64,ds=1    -> [B,H,L,64]   (Q, K)
//   ls=1, ds=2048 -> [B,H,64,L]   (V transposed)
// ---------------------------------------------------------------------------
__global__ __launch_bounds__(128)
void qkv_proj_kernel(const __bf16* __restrict__ Xb, const __bf16* __restrict__ Wb,
                     const float* __restrict__ bias, __bf16* __restrict__ Y,
                     float scale, int ls, int ds)
{
  const int lane = threadIdx.x & 31;
  const int wave = threadIdx.x >> 5;
  const int hgrp = lane >> 4;                 // half-wave group
  const int ln   = lane & 15;

  const int tile = blockIdx.x * 4 + wave;     // 8192 wave-tiles
  const int tm = tile >> 5;                   // row-tile (0..255)
  const int tc = tile & 31;                   // 32-wide col-tile (0..31)
  const int row  = tm * 16 + ln;
  const int col0 = tc * 32 + ln;
  const int col1 = tc * 32 + 16 + ln;

  const __bf16* xr  = Xb + (size_t)row  * HIDDEN;
  const __bf16* w0r = Wb + (size_t)col0 * HIDDEN;
  const __bf16* w1r = Wb + (size_t)col1 * HIDDEN;

  v8f acc0 = {}, acc1 = {};
  for (int k0 = 0; k0 < HIDDEN; k0 += 32) {
    // A-layout: e -> k = 16*hgrp + e (contiguous, matches B)
    v8bf lo = *(const v8bf*)(xr + k0 + 16 * hgrp);
    v8bf hi = *(const v8bf*)(xr + k0 + 16 * hgrp + 8);
    v16bf a = __builtin_shufflevector(lo, hi, 0, 1, 2, 3, 4, 5, 6, 7,
                                      8, 9, 10, 11, 12, 13, 14, 15);
    // B-layout: e -> k = 16*hgrp + e (contiguous along W row)
    v16bf b0 = *(const v16bf*)(w0r + k0 + 16 * hgrp);
    v16bf b1 = *(const v16bf*)(w1r + k0 + 16 * hgrp);
    acc0 = wmma_bf16(a, b0, acc0);
    acc1 = wmma_bf16(a, b1, acc1);
  }

#pragma unroll
  for (int sub = 0; sub < 2; ++sub) {
    const int col = sub ? col1 : col0;
    const v8f acc = sub ? acc1 : acc0;
    const float bval = bias[col];
    const int h = col >> 6;
    const int d = col & 63;
#pragma unroll
    for (int v = 0; v < 8; ++v) {             // C/D layout: elem v -> row v+8*hgrp
      const int m  = tm * 16 + v + 8 * hgrp;
      const int bb = m >> 11;                 // batch = m / 2048
      const int l  = m & 2047;
      const float val = (acc[v] + bval) * scale;
      Y[(size_t)(bb * NH + h) * HEAD_ELEMS + (size_t)l * ls + (size_t)d * ds] =
          (__bf16)val;
    }
  }
}

// ---------------------------------------------------------------------------
// Flash attention: one block = (b, h, 128 q-rows); 8 waves x 16 q-rows each.
// K (32x64) and V^T (64x32) tiles are staged into LDS by the TDM (wave 0
// issues two tensor_load_to_lds per stage), double-buffered; all 8 waves
// share them. Per 32 k-columns: 4 WMMAs for S, online softmax, LDS transpose
// of P into A-fragment layout, 4 WMMAs for PV.
// ---------------------------------------------------------------------------
__global__ __launch_bounds__(256)
void flash_attn_kernel(const __bf16* __restrict__ Qg, const __bf16* __restrict__ Kg,
                       const __bf16* __restrict__ Vtg, const float* __restrict__ mask,
                       float* __restrict__ out)
{
  __shared__ __align__(16) __bf16 Kt [2][32 * 64];   // [klocal][d]  2 x 4 KB
  __shared__ __align__(16) __bf16 Vts[2][64 * 32];   // [d][klocal]  2 x 4 KB
  __shared__ __align__(16) __bf16 Plds[8][16 * 32];  // per-wave P    8 KB

  const int tid  = threadIdx.x;
  const int lane = tid & 31;
  const int wave = tid >> 5;
  const int hgrp = lane >> 4;
  const int ln   = lane & 15;

  const int idx  = blockIdx.x;               // 512 blocks
  const int qblk = idx & 15;
  const int h    = (idx >> 4) & 15;
  const int b    = idx >> 8;

  const size_t head = (size_t)(b * NH + h) * HEAD_ELEMS;
  const __bf16* Qh = Qg + head;              // [L, 64]
  const __bf16* Kh = Kg + head;              // [L, 64]
  const __bf16* Vh = Vtg + head;             // [64, L]
  const float*  mb = mask + (size_t)b * SEQ * SEQ;

  const int qb = qblk * 128 + wave * 16;     // this wave's 16 q-rows
  const int qrow_lane = qb + ln;

  // Q A-fragments for d=[0,64): two 16x32 fragments, resident in VGPRs.
  v16bf aq[2];
#pragma unroll
  for (int ks = 0; ks < 2; ++ks) {
    v8bf lo = *(const v8bf*)(Qh + (size_t)qrow_lane * HD + 32 * ks + 16 * hgrp);
    v8bf hi = *(const v8bf*)(Qh + (size_t)qrow_lane * HD + 32 * ks + 16 * hgrp + 8);
    aq[ks] = __builtin_shufflevector(lo, hi, 0, 1, 2, 3, 4, 5, 6, 7,
                                     8, 9, 10, 11, 12, 13, 14, 15);
  }

  // LDS byte offsets of the staging buffers (flat addr [31:0] == LDS offset).
  const unsigned kbase0 = (unsigned)(uintptr_t)(&Kt[0][0]);
  const unsigned kbase1 = (unsigned)(uintptr_t)(&Kt[1][0]);
  const unsigned vbase0 = (unsigned)(uintptr_t)(&Vts[0][0]);
  const unsigned vbase1 = (unsigned)(uintptr_t)(&Vts[1][0]);

  // TDM D# group1 descriptors (compile-time constants, live in SGPRs):
  //  K tile : rows of 64 bf16, 32 rows, row stride 64   -> LDS [32][64]
  //  Vt tile: rows of 32 bf16, 64 rows, row stride 2048 -> LDS [64][32]
  const v8u g1K = TDM_G1(/*tdim0=*/64, /*tdim1=*/32, /*tile0=*/64, /*tile1=*/32,
                         /*stride0=*/64);
  const v8u g1V = TDM_G1(/*tdim0=*/32, /*tdim1=*/64, /*tile0=*/32, /*tile1=*/64,
                         /*stride0=*/SEQ);

  // Stage the 32-k-column tile starting at k0 into buffer buf (wave 0 only).
  auto issue_stage = [&](int k0, int buf) {
    tdm_load_2d(buf ? kbase1 : kbase0, Kh + (size_t)k0 * HD, g1K);
    tdm_load_2d(buf ? vbase1 : vbase0, Vh + (size_t)k0, g1V);
  };

  float m_i[8], l_i[8];
  v8f o_acc[4];
#pragma unroll
  for (int v = 0; v < 8; ++v) { m_i[v] = -1e30f; l_i[v] = 0.0f; }
#pragma unroll
  for (int t = 0; t < 4; ++t) { v8f z = {}; o_acc[t] = z; }

  __bf16* Pl = &Plds[wave][0];

  if (wave == 0) {
    issue_stage(0, 0);
    __builtin_amdgcn_s_wait_tensorcnt(0);
  }
  __syncthreads();

  for (int k0 = 0; k0 < SEQ; k0 += 32) {
    const int buf = (k0 >> 5) & 1;
    if (wave == 0 && k0 + 32 < SEQ)
      issue_stage(k0 + 32, buf ^ 1);         // DMA overlaps with compute below

    const __bf16* Kl = &Kt[buf][0];
    const __bf16* Vl = &Vts[buf][0];
    const int kc0 = k0 + ln;                 // global column of S-tile 0
    const int kc1 = k0 + 16 + ln;            // global column of S-tile 1

    // ---- S = Q * K^T : two 16x16 tiles over 32 k-columns (from LDS) ------
    v8f s0 = {}, s1 = {};
#pragma unroll
    for (int ks = 0; ks < 2; ++ks) {
      v16bf bk0 = *(const v16bf*)(Kl + (size_t)ln * HD + 32 * ks + 16 * hgrp);
      v16bf bk1 = *(const v16bf*)(Kl + (size_t)(16 + ln) * HD + 32 * ks + 16 * hgrp);
      s0 = wmma_bf16(aq[ks], bk0, s0);
      s1 = wmma_bf16(aq[ks], bk1, s1);
    }

    // ---- mask + online softmax ------------------------------------------
    float p0[8], p1[8], rs[8];
#pragma unroll
    for (int v = 0; v < 8; ++v) {
      const int qr = qb + v + 8 * hgrp;      // C/D row for element v
      const float mk0 = mb[(size_t)qr * SEQ + kc0];
      const float mk1 = mb[(size_t)qr * SEQ + kc1];
      const float sv0 = s0[v] + (1.0f - mk0) * -10000.0f;
      const float sv1 = s1[v] + (1.0f - mk1) * -10000.0f;

      float mp = fmaxf(sv0, sv1);            // row-max across the 16-lane half
      mp = fmaxf(mp, __shfl_xor(mp, 1, 32));
      mp = fmaxf(mp, __shfl_xor(mp, 2, 32));
      mp = fmaxf(mp, __shfl_xor(mp, 4, 32));
      mp = fmaxf(mp, __shfl_xor(mp, 8, 32));
      const float mn = fmaxf(m_i[v], mp);

      const float e0 = __expf(sv0 - mn);
      const float e1 = __expf(sv1 - mn);
      float psum = e0 + e1;                  // row-sum across the 16-lane half
      psum += __shfl_xor(psum, 1, 32);
      psum += __shfl_xor(psum, 2, 32);
      psum += __shfl_xor(psum, 4, 32);
      psum += __shfl_xor(psum, 8, 32);

      rs[v]  = __expf(m_i[v] - mn);
      l_i[v] = l_i[v] * rs[v] + psum;
      m_i[v] = mn;
      p0[v] = e0;
      p1[v] = e1;
    }
#pragma unroll
    for (int t = 0; t < 4; ++t)
#pragma unroll
      for (int v = 0; v < 8; ++v) o_acc[t][v] *= rs[v];

    // ---- P: C/D layout -> LDS 16x32 -> reload as A fragment --------------
    // Same-wave DS ops are in-order, so no barrier is needed here.
#pragma unroll
    for (int v = 0; v < 8; ++v) {
      Pl[(v + 8 * hgrp) * 32 + ln]      = (__bf16)p0[v];
      Pl[(v + 8 * hgrp) * 32 + 16 + ln] = (__bf16)p1[v];
    }
    v8bf plo = *(const v8bf*)(Pl + ln * 32 + 16 * hgrp);
    v8bf phi = *(const v8bf*)(Pl + ln * 32 + 16 * hgrp + 8);
    v16bf ap = __builtin_shufflevector(plo, phi, 0, 1, 2, 3, 4, 5, 6, 7,
                                       8, 9, 10, 11, 12, 13, 14, 15);

    // ---- O += P * V : Vt tile rows (fixed d) contiguous in k (from LDS) --
#pragma unroll
    for (int t = 0; t < 4; ++t) {
      v16bf bv = *(const v16bf*)(Vl + (size_t)(t * 16 + ln) * 32 + 16 * hgrp);
      o_acc[t] = wmma_bf16(ap, bv, o_acc[t]);
    }

    // Next stage must be resident (and this stage fully consumed) before the
    // buffers are reused: wave 0 drains TENSORcnt, then workgroup barrier.
    if (wave == 0) __builtin_amdgcn_s_wait_tensorcnt(0);
    __syncthreads();
  }

  // ---- epilogue: normalize, write [B, LQ, H*D] f32 ------------------------
#pragma unroll
  for (int v = 0; v < 8; ++v) {
    const int qr = qb + v + 8 * hgrp;
    const float inv = 1.0f / l_i[v];
    float* orow = out + ((size_t)b * SEQ + qr) * HIDDEN + h * HD;
#pragma unroll
    for (int t = 0; t < 4; ++t) orow[t * 16 + ln] = o_acc[t][v] * inv;
  }
}

// ---------------------------------------------------------------------------
extern "C" void kernel_launch(void* const* d_in, const int* in_sizes, int n_in,
                              void* d_out, int out_size, void* d_ws, size_t ws_size,
                              hipStream_t stream) {
  (void)in_sizes; (void)n_in; (void)out_size;

  const float* Xq   = (const float*)d_in[0];   // [2,2048,1024]
  const float* Xk   = (const float*)d_in[1];   // [2,2048,1024]
  const float* mask = (const float*)d_in[2];   // [2,2048,2048]
  const float* Wq   = (const float*)d_in[3];
  const float* bq   = (const float*)d_in[4];
  const float* Wk   = (const float*)d_in[5];
  const float* bk   = (const float*)d_in[6];
  const float* Wv   = (const float*)d_in[7];
  const float* bv   = (const float*)d_in[8];

  // Workspace: Q | K | Vt | Xq_bf | Xk_bf | Wq_bf | Wk_bf | Wv_bf  (~46 MB)
  const size_t WELEMS = (size_t)HIDDEN * HIDDEN;          // 1,048,576
  const size_t need = (5 * (size_t)QKV_ELEMS + 3 * WELEMS) * sizeof(__bf16);
  if (ws_size < need) return;

  __bf16* Q   = (__bf16*)d_ws;
  __bf16* K   = Q + QKV_ELEMS;
  __bf16* Vt  = K + QKV_ELEMS;
  __bf16* Xqb = Vt + QKV_ELEMS;        // [4096,1024] bf16 (== QKV_ELEMS elems)
  __bf16* Xkb = Xqb + QKV_ELEMS;
  __bf16* Wqb = Xkb + QKV_ELEMS;
  __bf16* Wkb = Wqb + WELEMS;
  __bf16* Wvb = Wkb + WELEMS;

  // One-time bf16 conversion of activations and weights.
  cvt_f32_bf16_kernel<<<dim3(4096), dim3(256), 0, stream>>>(Xq, Xqb, QKV_ELEMS / 4);
  cvt_f32_bf16_kernel<<<dim3(4096), dim3(256), 0, stream>>>(Xk, Xkb, QKV_ELEMS / 4);
  cvt_f32_bf16_kernel<<<dim3(1024), dim3(256), 0, stream>>>(Wq, Wqb, (int)(WELEMS / 4));
  cvt_f32_bf16_kernel<<<dim3(1024), dim3(256), 0, stream>>>(Wk, Wkb, (int)(WELEMS / 4));
  cvt_f32_bf16_kernel<<<dim3(1024), dim3(256), 0, stream>>>(Wv, Wvb, (int)(WELEMS / 4));

  // Projections (16x32 tile per wave, 4 waves per block: 2048 blocks).
  const dim3 pgrid(2048), pblk(128);
  qkv_proj_kernel<<<pgrid, pblk, 0, stream>>>(Xqb, Wqb, bq, Q, 0.125f, HD, 1);
  qkv_proj_kernel<<<pgrid, pblk, 0, stream>>>(Xkb, Wkb, bk, K, 1.0f, HD, 1);
  qkv_proj_kernel<<<pgrid, pblk, 0, stream>>>(Xkb, Wvb, bv, Vt, 1.0f, 1, SEQ);

  flash_attn_kernel<<<dim3(512), dim3(256), 0, stream>>>(Q, K, Vt, mask,
                                                         (float*)d_out);
}
